// GaussianDPMM_20959440405087
// MI455X (gfx1250) — matrix operand order repaired and checked by reading, not compiled, not run on hardware
//
#include <hip/hip_runtime.h>
#include <cstdint>
#include <cstddef>

#define NPTS 262144
#define KCL  256
#define DDIM 64
#define FDIM 128
#define LOG2PI_F 1.8378770664093453f
#define LOG2_F   0.6931471805599453f

typedef __attribute__((ext_vector_type(16))) _Float16 v16h;
typedef __attribute__((ext_vector_type(8)))  float    v8f;

// ---------- device math helpers ----------
__device__ __forceinline__ float digammaf_(float x) {
    // recurrence to x >= 6, then asymptotic series
    float r = 0.0f;
    while (x < 6.0f) { r -= 1.0f / x; x += 1.0f; }
    float inv  = 1.0f / x;
    float inv2 = inv * inv;
    return r + logf(x) - 0.5f * inv
         - inv2 * (1.0f / 12.0f - inv2 * (1.0f / 120.0f - inv2 * (1.0f / 252.0f)));
}

__device__ __forceinline__ unsigned int packh2(_Float16 lo16, _Float16 hi16) {
    unsigned short a, b;
    __builtin_memcpy(&a, &lo16, 2);
    __builtin_memcpy(&b, &hi16, 2);
    return (unsigned int)a | ((unsigned int)b << 16);
}

// =====================================================================
// Kernel 1: per-cluster prep. Computes elog_pi (stick-breaking scan),
// per-cluster bias const[k], all KL terms (-> elbo_const), and the GEMM
// weights W (128 x 256) split into hi/lo f16 planes written directly in
// contiguous-k B-fragment order, same as A (lane = h*16 + col%16, word j holds K = h*16+2j,+1).
// =====================================================================
__global__ __launch_bounds__(KCL)
void dpmm_prep(const float* __restrict__ u,   const float* __restrict__ v,
               const float* __restrict__ tau, const float* __restrict__ c,
               const float* __restrict__ n,   const float* __restrict__ B,
               const float* __restrict__ tau0,const float* __restrict__ c0,
               const float* __restrict__ n0,  const float* __restrict__ B0,
               unsigned int* __restrict__ wfrag,   // 2 planes * 16384 u32
               float* __restrict__ constk,         // 256
               float* __restrict__ elbo_const)     // 1
{
    __shared__ float sh_e1b[KCL];
    __shared__ float sh_excl[KCL];
    __shared__ float sh_red[KCL];

    const int k = threadIdx.x;
    const float uk = u[k], vk = v[k], ck = c[k], nk = n[k];
    const float c0k = c0[k], n0k = n0[k];

    const float dsum = digammaf_(uk + vk);
    const float eb   = digammaf_(uk) - dsum;   // E[log beta]
    const float e1b  = digammaf_(vk) - dsum;   // E[log(1-beta)]

    sh_e1b[k] = e1b;
    __syncthreads();
    if (k == 0) {                               // exclusive cumsum (K=256, trivial)
        float a = 0.0f;
        for (int i = 0; i < KCL; ++i) { sh_excl[i] = a; a += sh_e1b[i]; }
    }
    __syncthreads();
    const float elogpi = eb + sh_excl[k];

    const float dg_halfn = digammaf_(0.5f * nk);
    float S = 0.0f, t2 = 0.0f, klg = 0.0f, knq = 0.0f;
    for (int d = 0; d < DDIM; ++d) {
        const float Bkd  = B [k * DDIM + d];
        const float tkd  = tau [k * DDIM + d];
        const float B0kd = B0[k * DDIM + d];
        const float t0kd = tau0[k * DDIM + d];
        const float El   = nk / Bkd;
        S  += dg_halfn + LOG2_F - logf(Bkd);
        t2 += El * tkd * tkd;
        const float a  = 0.5f * nk,  a0 = 0.5f * n0k;
        const float b  = 0.5f * Bkd, b0 = 0.5f * B0kd;
        klg += (a - a0) * digammaf_(a) - lgammaf(a) + lgammaf(a0)
             + a0 * (logf(b) - logf(b0)) + a * (b0 - b) / b;
        const float dt = tkd - t0kd;
        knq += El * dt * dt;
    }
    constk[k] = elogpi + 0.5f * (S - (float)DDIM * LOG2PI_F - (float)DDIM / ck - t2);

    // ---- W fragments: feature f<64 -> coeff of x^2 = -0.5*Elam
    //                   feature f>=64 -> coeff of x   = Elam*tau
    const int ct   = k >> 4;
    const int nlow = k & 15;
    for (int p = 0; p < FDIM / 2; ++p) {
        const int fe = 2 * p;                        // even feature index
        float w0, w1;
        if (fe < DDIM) {
            w0 = -0.5f * (nk / B[k * DDIM + fe]);
            w1 = -0.5f * (nk / B[k * DDIM + fe + 1]);
        } else {
            const int d = fe - DDIM;
            w0 = (nk / B[k * DDIM + d])     * tau[k * DDIM + d];
            w1 = (nk / B[k * DDIM + d + 1]) * tau[k * DDIM + d + 1];
        }
        const _Float16 h0 = (_Float16)w0, h1 = (_Float16)w1;
        const _Float16 l0 = (_Float16)(w0 - (float)h0);
        const _Float16 l1 = (_Float16)(w1 - (float)h1);

        // B operand (32x16 per k-step), contiguous k as in A: lane = h*16 + col, VGPR j: K = h*16 + 2j,+1
        const int ks   = fe >> 5;
        const int r    = fe & 31;
        const int hh   = r >> 4;
        const int j    = (r & 15) >> 1;
        const int lane = hh * 16 + nlow;
        const int base = ((ks * 16 + ct) * 32 + lane) * 8 + j;
        wfrag[base]          = packh2(h0, h1);   // hi plane
        wfrag[16384 + base]  = packh2(l0, l1);   // lo plane
    }

    // ---- KL terms -> elbo constant
    const float klb = lgammaf(uk + vk) - lgammaf(uk) - lgammaf(vk)
                    + (uk - 1.0f) * eb + (vk - 1.0f) * e1b;      // ALPHA = 1
    const float kn  = 0.5f * (float)DDIM * (c0k / ck - 1.0f + logf(ck / c0k))
                    + 0.5f * c0k * knq;
    sh_red[k] = klb + klg + kn;
    __syncthreads();
    if (k == 0) {
        float s = 0.0f;
        for (int i = 0; i < KCL; ++i) s += sh_red[i];
        elbo_const[0] = -s;
    }
}

// =====================================================================
// Kernel 2: fused split-f16 WMMA GEMM (Phi @ W) + bias + row logsumexp
// + softmax. Block = 8 wave32, 128 rows; each wave owns 16 rows x all
// 256 columns (16 accumulator tiles), so the softmax is wave-local.
// =====================================================================
__global__ __launch_bounds__(256)
void dpmm_main(const float* __restrict__ x,
               const unsigned int* __restrict__ wfrag,
               const float* __restrict__ constk,
               float* __restrict__ pi,
               float* __restrict__ partials)
{
    extern __shared__ unsigned int smem_u32[];
    unsigned int* wsm  = smem_u32;                  // 32768 u32 = 128 KB (hi+lo planes)
    float*        csm  = (float*)(smem_u32 + 32768);// 256 bias values
    float*        wred = csm + KCL;                 // 8 per-wave lse partials

    const int tid = threadIdx.x;

    // stage W fragments + bias into LDS (coalesced uint4)
    {
        const uint4* src = (const uint4*)wfrag;
        uint4*       dst = (uint4*)wsm;
        for (int i = tid; i < 8192; i += 256) dst[i] = src[i];
        csm[tid] = constk[tid];
    }
    __syncthreads();

    const int wid  = tid >> 5;
    const int lane = tid & 31;
    const int hh   = lane >> 4;       // lane half
    const int m    = lane & 15;       // A row within tile / C column within tile
    const int row  = blockIdx.x * 128 + wid * 16 + m;
    const float4* xr = (const float4*)(x + (size_t)row * DDIM);

    v8f acc[16];
    #pragma unroll
    for (int ct = 0; ct < 16; ++ct) {
        #pragma unroll
        for (int j = 0; j < 8; ++j) acc[ct][j] = 0.0f;
    }

    #pragma unroll
    for (int ks = 0; ks < 4; ++ks) {
        // A operand 16x32 f16, contiguous k (same as B): lane(0-15)=row, VGPR j: K = h*16 + 2*j
        const int  ksd = ks & 1;                  // which 32-dim block of x
        const bool sq  = (ks < 2);                // x^2 features first, then x
        const int  d0  = ksd * 32 + hh * 16;      // chunk for VGPRs 0..3
        const int  d1  = d0 + 8;                  // chunk for VGPRs 4..7
        const float4 a0 = xr[(d0 >> 2) + 0];
        const float4 a1 = xr[(d0 >> 2) + 1];
        const float4 b0 = xr[(d1 >> 2) + 0];
        const float4 b1 = xr[(d1 >> 2) + 1];
        const float fv[16] = { a0.x, a0.y, a0.z, a0.w,  a1.x, a1.y, a1.z, a1.w,
                               b0.x, b0.y, b0.z, b0.w,  b1.x, b1.y, b1.z, b1.w };
        v16h ahi, alo;
        #pragma unroll
        for (int e = 0; e < 16; ++e) {
            const float val = sq ? fv[e] * fv[e] : fv[e];
            const _Float16 hi = (_Float16)val;
            ahi[e] = hi;
            alo[e] = (_Float16)(val - (float)hi);
        }

        #pragma unroll
        for (int ct = 0; ct < 16; ++ct) {
            const int fb = ((ks * 16 + ct) * 32 + lane) * 8;
            const v16h bhi = *(const v16h*)(wsm + fb);
            const v16h blo = *(const v16h*)(wsm + 16384 + fb);
            // (Ah+Al)(Bh+Bl) ~= AhBh + AhBl + AlBh, fp32 accumulate
            acc[ct] = __builtin_amdgcn_wmma_f32_16x16x32_f16(false, ahi, false, bhi,
                                                             (short)0, acc[ct], false, false);
            acc[ct] = __builtin_amdgcn_wmma_f32_16x16x32_f16(false, ahi, false, blo,
                                                             (short)0, acc[ct], false, false);
            acc[ct] = __builtin_amdgcn_wmma_f32_16x16x32_f16(false, alo, false, bhi,
                                                             (short)0, acc[ct], false, false);
        }
    }

    // ---- epilogue: bias, row max, exp-sum, softmax, lse ----
    #pragma unroll
    for (int ct = 0; ct < 16; ++ct) {
        const float bias = csm[ct * 16 + m];      // column = ct*16 + m
        #pragma unroll
        for (int j = 0; j < 8; ++j) acc[ct][j] += bias;
    }

    float mx[8];
    #pragma unroll
    for (int j = 0; j < 8; ++j) {                 // row = wave_base + j + 8*hh
        float t = acc[0][j];
        #pragma unroll
        for (int ct = 1; ct < 16; ++ct) t = fmaxf(t, acc[ct][j]);
        t = fmaxf(t, __shfl_xor(t, 1));
        t = fmaxf(t, __shfl_xor(t, 2));
        t = fmaxf(t, __shfl_xor(t, 4));
        t = fmaxf(t, __shfl_xor(t, 8));           // stays within 16-lane half
        mx[j] = t;
    }

    float sm[8];
    #pragma unroll
    for (int j = 0; j < 8; ++j) sm[j] = 0.0f;
    #pragma unroll
    for (int ct = 0; ct < 16; ++ct) {
        #pragma unroll
        for (int j = 0; j < 8; ++j) {
            const float e = __expf(acc[ct][j] - mx[j]);
            acc[ct][j] = e;                       // reuse acc as exp storage
            sm[j] += e;
        }
    }

    float lsesum = 0.0f;
    float rs[8];
    #pragma unroll
    for (int j = 0; j < 8; ++j) {
        float t = sm[j];
        t += __shfl_xor(t, 1);
        t += __shfl_xor(t, 2);
        t += __shfl_xor(t, 4);
        t += __shfl_xor(t, 8);
        rs[j] = 1.0f / t;
        lsesum += mx[j] + logf(t);
    }

    const int rb = blockIdx.x * 128 + wid * 16 + hh * 8;
    #pragma unroll
    for (int ct = 0; ct < 16; ++ct) {
        #pragma unroll
        for (int j = 0; j < 8; ++j) {
            pi[(size_t)(rb + j) * KCL + ct * 16 + m] = acc[ct][j] * rs[j];
        }
    }

    // deterministic ELBO partial: lsesum is uniform within each 16-lane half
    const float tot = lsesum + __shfl_xor(lsesum, 16);   // 16 rows of this wave
    if (lane == 0) wred[wid] = tot;
    __syncthreads();
    if (tid == 0) {
        float s = 0.0f;
        for (int i = 0; i < 8; ++i) s += wred[i];
        partials[blockIdx.x] = s;
    }
}

// =====================================================================
// Kernel 3: fixed-order reduction of 2048 block partials + KL constant
// =====================================================================
__global__ __launch_bounds__(256)
void dpmm_finish(const float* __restrict__ partials,
                 const float* __restrict__ elbo_const,
                 float* __restrict__ out_elbo)
{
    __shared__ float red[256];
    const int t = threadIdx.x;
    float s = 0.0f;
    for (int i = t; i < 2048; i += 256) s += partials[i];
    red[t] = s;
    __syncthreads();
    for (int off = 128; off > 0; off >>= 1) {
        if (t < off) red[t] += red[t + off];
        __syncthreads();
    }
    if (t == 0) out_elbo[0] = elbo_const[0] + red[0];
}

// =====================================================================
extern "C" void kernel_launch(void* const* d_in, const int* in_sizes, int n_in,
                              void* d_out, int out_size, void* d_ws, size_t ws_size,
                              hipStream_t stream) {
    const float* x    = (const float*)d_in[0];
    const float* u    = (const float*)d_in[1];
    const float* v    = (const float*)d_in[2];
    const float* tau  = (const float*)d_in[3];
    const float* c    = (const float*)d_in[4];
    const float* n    = (const float*)d_in[5];
    const float* B    = (const float*)d_in[6];
    const float* tau0 = (const float*)d_in[7];
    const float* c0   = (const float*)d_in[8];
    const float* n0   = (const float*)d_in[9];
    const float* B0   = (const float*)d_in[10];

    float* out  = (float*)d_out;
    float* pi   = out;                                  // N*K
    float* elbo = out + (size_t)NPTS * KCL;             // scalar

    // workspace layout
    unsigned int* wfrag    = (unsigned int*)d_ws;                       // 131072 B
    float*        constk   = (float*)((char*)d_ws + 131072);            // 1024 B
    float*        elbo_c   = (float*)((char*)d_ws + 132096);            // 4 B
    float*        partials = (float*)((char*)d_ws + 132352);            // 8192 B

    dpmm_prep<<<1, KCL, 0, stream>>>(u, v, tau, c, n, B, tau0, c0, n0, B0,
                                     wfrag, constk, elbo_c);

    const size_t smem = 32768u * 4u + KCL * 4u + 8u * 4u;   // 132128 B
    dpmm_main<<<NPTS / 128, 256, smem, stream>>>(x, wfrag, constk, pi, partials);

    dpmm_finish<<<1, 256, 0, stream>>>(partials, elbo_c, elbo);
}
